// GraphQNN__gen108_65481071401394
// MI455X (gfx1250) — matrix operand order repaired and checked by reading, not compiled, not run on hardware
//
#include <hip/hip_runtime.h>
#include <hip/hip_bf16.h>
#include <math.h>

typedef _Float16 half_t;
typedef __attribute__((ext_vector_type(16))) _Float16 v16h;
typedef __attribute__((ext_vector_type(8)))  _Float16 v8h;
typedef __attribute__((ext_vector_type(4)))  _Float16 v4h;
typedef __attribute__((ext_vector_type(8)))  float    v8f;
typedef __attribute__((ext_vector_type(2)))  _Float16 h2;
typedef __attribute__((ext_vector_type(4)))  int      i4;

#define NQ   16
#define SDIM 65536

#ifndef __has_builtin
#define __has_builtin(x) 0
#endif
#if __has_builtin(__builtin_amdgcn_global_load_async_to_lds_b128)
#define ASYNC_LDS_BUILTIN 1
#endif
#if __has_builtin(__builtin_amdgcn_s_wait_asynccnt)
#define WAIT_ASYNC_BUILTIN 1
#endif

// pointer types matching the builtin's expected signature (int4* in AS1/AS3)
typedef __attribute__((address_space(1))) i4 gi4_t;
typedef __attribute__((address_space(3))) i4 li4_t;

// async 16-byte global -> LDS copy (gfx1250 GLOBAL_LOAD_ASYNC_TO_LDS_B128)
__device__ __forceinline__ void async_copy_b128(const half_t* g, half_t* l) {
#ifdef ASYNC_LDS_BUILTIN
    gi4_t* gp = (gi4_t*)(unsigned long long)(uintptr_t)g;    // global AS, via integer
    li4_t* lp = (li4_t*)(unsigned)(uintptr_t)l;              // LDS offset = low 32 bits
    __builtin_amdgcn_global_load_async_to_lds_b128(gp, lp, 0, 0);
#else
    asm volatile("global_load_async_to_lds_b128 %0, %1, off"
                 :: "v"((unsigned)(uintptr_t)l),
                    "v"((unsigned long long)(uintptr_t)g)
                 : "memory");
#endif
}
__device__ __forceinline__ void wait_async0() {
#ifdef WAIT_ASYNC_BUILTIN
    __builtin_amdgcn_s_wait_asynccnt(0);
#else
    asm volatile("s_wait_asynccnt 0x0" ::: "memory");
#endif
}

// ---------------- f32 -> f16 convert (for weights) ----------------
__global__ void cvt_f32_f16(const float* __restrict__ in, half_t* __restrict__ out, int n) {
    int i = blockIdx.x * blockDim.x + threadIdx.x;
    if (i < n) out[i] = (half_t)in[i];
}

// ---------------- fused GEMM + bias + tanh ----------------
// C[M,N] = tanh(A[M,K] @ B[K,N] + bias[N]);  A f32 (AF32) or f16, B f16, C f16.
// Block: 256 threads = 8 waves; block tile 128(M) x 128(N), K step 32.
// Wave (wm 0..3, wn 0..1) owns a 32x64 strip = 2x4 WMMA accum fragments.
template<bool AF32>
__global__ __launch_bounds__(256)
void gemm_tanh(const void* __restrict__ Ap, const half_t* __restrict__ B,
               const float* __restrict__ bias, half_t* __restrict__ C,
               int M, int N, int K)
{
    __shared__ half_t sA[128 * 40];    // 128 rows x 32 k, stride 40 (80B, 16B-aligned)
    __shared__ half_t sBT[128 * 40];   // 128 cols x 32 k (K-transposed), stride 40

    const int tid  = threadIdx.x;
    const int lane = tid & 31;
    const int wave = tid >> 5;
    const int wm   = wave >> 1;        // 0..3 (32-row strip)
    const int wn   = wave & 1;         // 0..1 (64-col strip)
    const int frow = lane & 15;
    const int hi   = lane >> 4;        // half-wave select per WMMA layout

    const int rowBase = blockIdx.y * 128;
    const int colBase = blockIdx.x * 128;

    v8f acc[2][4] = {};

    // cooperative tile-load coordinates
    const int ar  = tid >> 1;          // 0..127  (A row)
    const int ac  = (tid & 1) * 16;    // 0,16    (A col, 16 elems)
    const int bn  = (tid & 31) * 4;    // 0..124  (B col group of 4)
    const int bkk = (tid >> 5) * 4;    // 0..28   (B k group of 4)

    for (int k0 = 0; k0 < K; k0 += 32) {
        // ---- stage A tile (128x32) ----
        if (AF32) {
            const float*  A = (const float*)Ap;
            const float4* p = (const float4*)(A + (size_t)(rowBase + ar) * K + k0 + ac);
            float4 x0 = p[0], x1 = p[1], x2 = p[2], x3 = p[3];
            v8h h0 = { (half_t)x0.x, (half_t)x0.y, (half_t)x0.z, (half_t)x0.w,
                       (half_t)x1.x, (half_t)x1.y, (half_t)x1.z, (half_t)x1.w };
            v8h h1 = { (half_t)x2.x, (half_t)x2.y, (half_t)x2.z, (half_t)x2.w,
                       (half_t)x3.x, (half_t)x3.y, (half_t)x3.z, (half_t)x3.w };
            *(v8h*)&sA[ar * 40 + ac]     = h0;
            *(v8h*)&sA[ar * 40 + ac + 8] = h1;
            if (k0 + 32 < K)
                __builtin_prefetch(A + (size_t)(rowBase + ar) * K + k0 + 32 + ac, 0, 1);
        } else {
            const half_t* A = (const half_t*)Ap + (size_t)(rowBase + ar) * K + k0 + ac;
            async_copy_b128(A,     &sA[ar * 40 + ac]);
            async_copy_b128(A + 8, &sA[ar * 40 + ac + 8]);
        }
        // ---- stage B tile (32x128) K-transposed, 4x4 blocks per thread ----
        {
            const half_t* bp = B + (size_t)(k0 + bkk) * N + colBase + bn;
            v4h r0 = *(const v4h*)(bp);
            v4h r1 = *(const v4h*)(bp + N);
            v4h r2 = *(const v4h*)(bp + 2 * (size_t)N);
            v4h r3 = *(const v4h*)(bp + 3 * (size_t)N);
            #pragma unroll
            for (int i = 0; i < 4; ++i) {
                v4h cv = { r0[i], r1[i], r2[i], r3[i] };      // k-contiguous column slice
                *(v4h*)&sBT[(bn + i) * 40 + bkk] = cv;        // 8B-aligned b64 store
            }
        }
        if (!AF32) wait_async0();
        __syncthreads();

        // ---- A fragments (ISA 16-bit A layout: contiguous 8-half runs per half-wave) ----
        v16h af[2];
        #pragma unroll
        for (int mf = 0; mf < 2; ++mf) {
            int rr = wm * 32 + mf * 16 + frow;
            v8h a0 = *(const v8h*)&sA[rr * 40 + hi * 8];          // K 0-7 / 8-15
            v8h a1 = *(const v8h*)&sA[rr * 40 + 16 + hi * 8];     // K 16-23 / 24-31
            af[mf] = __builtin_shufflevector(a0, a1, 0,1,2,3,4,5,6,7,8,9,10,11,12,13,14,15);
        }
        // ---- B fragments: two 8-half k runs per lane (same k order as A) -> 2x ds_load_b128 ----
        #pragma unroll
        for (int nf = 0; nf < 4; ++nf) {
            int cc = wn * 64 + nf * 16 + frow;                    // N = lane&15
            v8h b0 = *(const v8h*)&sBT[cc * 40 + hi * 8];         // K 0-7 / 8-15
            v8h b1 = *(const v8h*)&sBT[cc * 40 + 16 + hi * 8];    // K 16-23 / 24-31
            v16h bf = __builtin_shufflevector(b0, b1, 0,1,2,3,4,5,6,7,8,9,10,11,12,13,14,15);
            acc[0][nf] = __builtin_amdgcn_wmma_f32_16x16x32_f16(
                false, af[0], false, bf, (short)0, acc[0][nf], false, false);
            acc[1][nf] = __builtin_amdgcn_wmma_f32_16x16x32_f16(
                false, af[1], false, bf, (short)0, acc[1][nf], false, false);
        }
        __syncthreads();
    }

    // ---- epilogue: bias + tanh, f16 store (C/D layout: VGPR v -> M=v+8*hi, N=lane&15) ----
    #pragma unroll
    for (int mf = 0; mf < 2; ++mf) {
        #pragma unroll
        for (int nf = 0; nf < 4; ++nf) {
            int col = colBase + wn * 64 + nf * 16 + frow;
            float bv = bias[col];
            #pragma unroll
            for (int v = 0; v < 8; ++v) {
                int r = rowBase + wm * 32 + mf * 16 + hi * 8 + v;
                C[(size_t)r * N + col] = (half_t)tanhf(acc[mf][nf][v] + bv);
            }
        }
    }
}

// ---------------- 16-qubit statevector circuit, state resident in LDS ----------------
// One 1024-thread workgroup per batch element. State: 65536 f16-complex = 256 KB LDS
// (fits CDNA5's 320 KB/WGP). Mean-pool embed, L2 normalize, DEPTH=2 circuit, Z expvals.
__global__ __launch_bounds__(1024)
void qnn_circuit(const half_t* __restrict__ H2, const float* __restrict__ qp,
                 float* __restrict__ out)
{
    extern __shared__ unsigned char smem[];
    h2*    state = (h2*)smem;                                   // 262144 B
    float* gtab  = (float*)(smem + SDIM * 4);                   // 32 gates * 8 f32
    float* red   = (float*)(smem + SDIM * 4 + 1024);            // 1024 f32
    float* red2  = (float*)(smem + SDIM * 4 + 1024 + 4096);     // 1024 f32

    const int t = threadIdx.x;
    const int b = blockIdx.x;

    // 1) fused gate matrices U = RZ(az) RY(ay) RX(ax), half-angles
    if (t < 32) {
        int d = t >> 4, q = t & 15;
        float ax = qp[d * 48 + q * 3 + 0] * 0.5f;
        float ay = qp[d * 48 + q * 3 + 1] * 0.5f;
        float az = qp[d * 48 + q * 3 + 2] * 0.5f;
        float cx = cosf(ax), sx = sinf(ax);
        float cy = cosf(ay), sy = sinf(ay);
        float cz = cosf(az), sz = sinf(az);
        float m00r =  cy * cx, m00i =  sy * sx;   // M = RY @ RX
        float m01r = -sy * cx, m01i = -cy * sx;
        float m10r =  sy * cx, m10i = -cy * sx;
        float m11r =  cy * cx, m11i = -sy * sx;
        float* g = &gtab[t * 8];
        g[0] = m00r * cz + m00i * sz;  g[1] = m00i * cz - m00r * sz;   // row0 * e^{-i az}
        g[2] = m01r * cz + m01i * sz;  g[3] = m01i * cz - m01r * sz;
        g[4] = m10r * cz - m10i * sz;  g[5] = m10i * cz + m10r * sz;   // row1 * e^{+i az}
        g[6] = m11r * cz - m11i * sz;  g[7] = m11i * cz + m11r * sz;
    }

    // 2) mean over N=1024 rows (coalesced across 512 features, two halves)
    {
        int f = t & 511, hsel = t >> 9;
        const half_t* base = H2 + ((size_t)b * 1024 + (size_t)hsel * 512) * 512 + f;
        float s = 0.f;
        for (int n = 0; n < 512; ++n) s += (float)base[(size_t)n * 512];
        red[t] = s;
    }
    __syncthreads();
    if (t < 512) red[t] = (red[t] + red[t + 512]) * (1.0f / 1024.0f);
    __syncthreads();

    // 3) L2 norm of embed
    red2[t] = (t < 512) ? red[t] * red[t] : 0.f;
    __syncthreads();
    for (int s = 512; s > 0; s >>= 1) {
        if (t < s) red2[t] += red2[t + s];
        __syncthreads();
    }
    float nrm = sqrtf(red2[0]);
    float inv = (nrm > 0.f) ? 1.0f / fmaxf(nrm, 1e-30f) : 1.0f;
    __syncthreads();

    // 4) init |psi>
    for (int i = t; i < SDIM; i += 1024) {
        h2 v;
        v[0] = (half_t)((i < 512) ? red[i] * inv : 0.f);
        v[1] = (half_t)0.f;
        state[i] = v;
    }
    __syncthreads();

    // 5) circuit: DEPTH layers of 16 1q gates + 15 adjacent CNOTs (wire 0 = MSB)
    for (int d = 0; d < 2; ++d) {
        for (int q = 0; q < NQ; ++q) {
            const float* g = &gtab[(d * 16 + q) * 8];
            float u00r = g[0], u00i = g[1], u01r = g[2], u01i = g[3];
            float u10r = g[4], u10i = g[5], u11r = g[6], u11i = g[7];
            int s   = 15 - q;
            int mlo = (1 << s) - 1;
            for (int p = t; p < 32768; p += 1024) {
                int i0 = ((p & ~mlo) << 1) | (p & mlo);
                int i1 = i0 | (1 << s);
                h2 a0 = state[i0], a1 = state[i1];
                float a0r = (float)a0[0], a0i = (float)a0[1];
                float a1r = (float)a1[0], a1i = (float)a1[1];
                float n0r = u00r * a0r - u00i * a0i + u01r * a1r - u01i * a1i;
                float n0i = u00r * a0i + u00i * a0r + u01r * a1i + u01i * a1r;
                float n1r = u10r * a0r - u10i * a0i + u11r * a1r - u11i * a1i;
                float n1i = u10r * a0i + u10i * a0r + u11r * a1i + u11i * a1r;
                h2 o0; o0[0] = (half_t)n0r; o0[1] = (half_t)n0i;
                h2 o1; o1[0] = (half_t)n1r; o1[1] = (half_t)n1i;
                state[i0] = o0;
                state[i1] = o1;
            }
            __syncthreads();
        }
        for (int q = 0; q < NQ - 1; ++q) {   // CNOT(q -> q+1): swap target where control=1
            int pt = 14 - q;
            int lo = (1 << pt) - 1;
            for (int p = t; p < 16384; p += 1024) {
                int i0 = ((p & ~lo) << 2) | (1 << (pt + 1)) | (p & lo);
                int i1 = i0 | (1 << pt);
                h2 a  = state[i0];
                h2 bb = state[i1];
                state[i0] = bb;
                state[i1] = a;
            }
            __syncthreads();
        }
    }

    // 6) Z expectation per qubit
    float ev[16];
    #pragma unroll
    for (int q = 0; q < 16; ++q) ev[q] = 0.f;
    for (int i = t; i < SDIM; i += 1024) {
        h2 a = state[i];
        float pr = (float)a[0], pi = (float)a[1];
        float p = pr * pr + pi * pi;
        #pragma unroll
        for (int q = 0; q < 16; ++q)
            ev[q] += ((i >> (15 - q)) & 1) ? -p : p;
    }
    for (int q = 0; q < 16; ++q) {
        __syncthreads();
        red[t] = ev[q];
        __syncthreads();
        for (int s = 512; s > 0; s >>= 1) {
            if (t < s) red[t] += red[t + s];
            __syncthreads();
        }
        if (t == 0) out[b * 16 + q] = red[0];
    }
}

extern "C" void kernel_launch(void* const* d_in, const int* in_sizes, int n_in,
                              void* d_out, int out_size, void* d_ws, size_t ws_size,
                              hipStream_t stream)
{
    const float* x  = (const float*)d_in[0];   // (64,1024,512)
    const float* W1 = (const float*)d_in[1];   // (512,1024)
    const float* b1 = (const float*)d_in[2];   // (1024,)
    const float* W2 = (const float*)d_in[3];   // (1024,512)
    const float* b2 = (const float*)d_in[4];   // (512,)
    const float* qp = (const float*)d_in[5];   // (96,)
    float* out = (float*)d_out;                // (64,16)

    const int F0 = 512, F1 = 1024, F2 = 512;
    const size_t Mrows = 65536;                // 64*1024

    char* ws = (char*)d_ws;
    half_t* C1h = (half_t*)ws;                                     // 128 MiB
    half_t* H2h = (half_t*)(ws + Mrows * F1 * 2);                  //  64 MiB
    half_t* W1h = (half_t*)(ws + Mrows * F1 * 2 + Mrows * F2 * 2); //   1 MiB
    half_t* W2h = (half_t*)((char*)W1h + (size_t)F0 * F1 * 2);     //   1 MiB

    cvt_f32_f16<<<(F0 * F1 + 255) / 256, 256, 0, stream>>>(W1, W1h, F0 * F1);
    cvt_f32_f16<<<(F1 * F2 + 255) / 256, 256, 0, stream>>>(W2, W2h, F1 * F2);

    dim3 g1(F1 / 128, (unsigned)(Mrows / 128));
    gemm_tanh<true><<<g1, 256, 0, stream>>>(x, W1h, b1, C1h, (int)Mrows, F1, F0);

    dim3 g2(F2 / 128, (unsigned)(Mrows / 128));
    gemm_tanh<false><<<g2, 256, 0, stream>>>(C1h, W2h, b2, H2h, (int)Mrows, F2, F1);

    size_t smem = (size_t)SDIM * 4 + 1024 + 4096 + 4096;  // 271360 B < 320 KB
    (void)hipFuncSetAttribute((const void*)qnn_circuit,
                              hipFuncAttributeMaxDynamicSharedMemorySize, (int)smem);
    qnn_circuit<<<64, 1024, smem, stream>>>(H2h, qp, out);
}
